// LSTM_37349035606683
// MI455X (gfx1250) — matrix operand order repaired and checked by reading, not compiled, not run on hardware
//
#include <hip/hip_runtime.h>
#include <hip/hip_bf16.h>
#include <math.h>

typedef __attribute__((ext_vector_type(16))) _Float16 v16h;
typedef __attribute__((ext_vector_type(8)))  _Float16 v8h;
typedef __attribute__((ext_vector_type(8)))  float    v8f;
typedef __attribute__((ext_vector_type(4)))  unsigned int u32x4;
typedef __attribute__((ext_vector_type(8)))  int i32x8;
typedef __attribute__((ext_vector_type(4)))  int i32x4;

#define D_IN 1024
#define HID  1024
#define KTOT 2048
#define SEQ  8192
#define NCLS 1000
#define NWG  128           // 64 WGs for block0 + 64 WGs for block1

#if defined(__has_builtin)
#  if __has_builtin(__builtin_amdgcn_tensor_load_to_lds) && \
      __has_builtin(__builtin_amdgcn_s_wait_tensorcnt)
#    define USE_TDM 1
#  endif
#endif

// ---------------------------------------------------------------------------
// Pack fp32 gate weights ([B][H][D+H], torch layout) into f16 WMMA B-fragment
// tiles. Per-lane B layout (wave32, 16-bit 32x16): lane L holds column N=L%16,
// K = 8*(L/16)+(e&7)+16*(e>>3).  Gate region = 2^21 f16; region order (block*4 + gate).
// Within a gate: tile (n_tile,k_tile) at (n_tile*64 + k_tile)*512 + lane*16+e.
// ---------------------------------------------------------------------------
__global__ void pack_weights(const float* __restrict__ Wf,
                             const float* __restrict__ Wi,
                             const float* __restrict__ Wc,
                             const float* __restrict__ Wo,
                             _Float16* __restrict__ wp) {
    unsigned id = blockIdx.x * 256u + threadIdx.x;      // < 16,777,216
    unsigned bg = id >> 21;
    unsigned g  = bg & 3u;
    unsigned b  = bg >> 2;
    unsigned s  = id & ((1u << 21) - 1u);
    unsigned tile  = s >> 9;
    unsigned li    = (s >> 4) & 31u;
    unsigned e     = s & 15u;
    unsigned ntile = tile >> 6;
    unsigned kt    = tile & 63u;
    unsigned row = ntile * 16u + (li & 15u);
    unsigned col = kt * 32u + ((li >> 4) << 3) + (e & 7u) + ((e >> 3) << 4);
    const float* W = (g == 0) ? Wf : (g == 1) ? Wi : (g == 2) ? Wc : Wo;
    float v = W[(size_t)b * HID * KTOT + (size_t)row * KTOT + col];
    wp[id] = (_Float16)v;
}

__global__ void convert_inputs(const float* __restrict__ x,
                               _Float16* __restrict__ xh) {
    unsigned id = blockIdx.x * 256u + threadIdx.x;      // < 8,388,608
    xh[id] = (_Float16)x[id];
}

// Zero h0/h1/o0h (3*2048 halves, contiguous), c (2048 f32), barrier words.
__global__ void init_state(_Float16* __restrict__ hz,
                           float* __restrict__ cz,
                           unsigned* __restrict__ bar) {
    unsigned id = blockIdx.x * 256u + threadIdx.x;
    if (id < 3u * 2u * HID) hz[id] = (_Float16)0.0f;
    if (id < 2u * HID)      cz[id] = 0.0f;
    if (blockIdx.x == 0 && threadIdx.x < 2) bar[threadIdx.x] = 0u;
}

// A-fragment: broadcast state chunk across all 16 rows.  16-bit A 16x32 ISA
// layout -> lane half (L/16) gives kbase in {0,8}; elems 0..7 = K kbase+0..7,
// elems 8..15 = K kbase+16..23 : two contiguous 16B f16 runs.
__device__ __forceinline__ v16h build_a(const _Float16* __restrict__ sp) {
    v8h r0 = *(const v8h*)(sp);
    v8h r1 = *(const v8h*)(sp + 16);
    return __builtin_shufflevector(r0, r1,
                                   0, 1, 2, 3, 4, 5, 6, 7,
                                   8, 9, 10, 11, 12, 13, 14, 15);
}

// ---------------------------------------------------------------------------
// Persistent LSTM kernel.  WG w: block b = w/64, j-tile jt = w%64 (16 hidden
// units).  4 waves = 4 gates.  Weights (256 KB) staged once into dynamic LDS
// (TDM tensor_load_to_lds when available), then 8193 pipelined phases:
// block0 computes step p while block1 computes step p-1; one grid barrier
// per phase (sense-reversing, agent-scope atomics).
// ---------------------------------------------------------------------------
__global__ void __launch_bounds__(128)
lstm_persistent(const _Float16* __restrict__ xh,
                const _Float16* __restrict__ wpack,
                _Float16* __restrict__ h0,      // [2][HID] ping-pong
                _Float16* __restrict__ h1,      // [2][HID] ping-pong
                _Float16* __restrict__ o0h,     // [2][HID] block0 o-gate (f16)
                float* __restrict__ cbuf,       // [2][HID]
                float* __restrict__ o1,         // [HID] block1 o-gate (f32)
                const float* __restrict__ bf_, const float* __restrict__ bi_,
                const float* __restrict__ bc_, const float* __restrict__ bo_,
                unsigned* __restrict__ bar_cnt, unsigned* __restrict__ bar_gen) {
    extern __shared__ char smem_raw[];
    _Float16* wl = (_Float16*)smem_raw;          // 4 gates * 32768 f16 = 256 KB
    __shared__ float gpre[4][16];

    const int tid   = threadIdx.x;
    const int wg    = blockIdx.x;
    const int bIdx  = wg >> 6;
    const int jt    = wg & 63;
    const int gate  = tid >> 5;
    const int lane  = tid & 31;
    const int kbase = (lane >> 4) << 3;
    const int j0    = jt * 16;

    // ---- stage this WG's weights into LDS (each wave loads its gate) ----
    {
        const _Float16* src = wpack + (((size_t)(bIdx * 4 + gate)) << 21)
                                    + ((size_t)jt << 15);
        _Float16* dst = wl + ((size_t)gate << 15);
#if defined(USE_TDM)
        // D# for a 64 KB contiguous tile: 2D tensor, 8192 x 1 of 8-byte elems.
        const unsigned lds_lo = (unsigned)(size_t)dst;   // low 32b = LDS addr
        const unsigned long long ga = (unsigned long long)(size_t)src;
        u32x4 g0 = { 1u,                                  // count=1, user mode
                     lds_lo,
                     (unsigned)(ga & 0xffffffffu),
                     (unsigned)((ga >> 32) & 0x01ffffffu) | (2u << 30) }; // type=2
        i32x8 g1 = { (int)(3u << 16),        // data_size = 8B (bits 17:16)
                     (int)(0x2000u << 16),   // tensor_dim0 lo16 = 8192 (bits 63:48)
                     (int)(1u << 16),        // tensor_dim1 = 1     (bits 95:80)
                     (int)(0x2000u << 16),   // tile_dim0 = 8192    (bits 127:112)
                     1,                      // tile_dim1 = 1       (bits 143:128)
                     0x2000,                 // tensor_dim0_stride = 8192
                     0, 0 };
        i32x4 gz = { 0, 0, 0, 0 };
#if __clang_major__ >= 23
        i32x8 gz8 = { 0, 0, 0, 0, 0, 0, 0, 0 };
        __builtin_amdgcn_tensor_load_to_lds(g0, g1, gz, gz, gz8, 0);
#else
        __builtin_amdgcn_tensor_load_to_lds(g0, g1, gz, gz, 0);
#endif
        __builtin_amdgcn_s_wait_tensorcnt(0);
#else
        for (int c = lane; c < 4096; c += 32)            // 4096 x 16B per gate
            *(v8h*)(dst + (size_t)c * 8) = *(const v8h*)(src + (size_t)c * 8);
#endif
    }
    __syncthreads();

    const float* bfv = bf_ + bIdx * HID;
    const float* biv = bi_ + bIdx * HID;
    const float* bcv = bc_ + bIdx * HID;
    const float* bov = bo_ + bIdx * HID;
    float* cb = cbuf + bIdx * HID;
    const _Float16* wt = wl + ((size_t)gate << 15) + (size_t)lane * 16;

    for (int p = 0; p <= SEQ; ++p) {
        const int t = (bIdx == 0) ? p : (p - 1);    // block1 runs one step behind
        if (t >= 0 && t < SEQ) {
            const _Float16* xv = (bIdx == 0) ? (xh + (size_t)t * D_IN)
                                             : (o0h + (size_t)(t & 1) * HID);
            const _Float16* hv = ((bIdx == 0) ? h0 : h1) + (size_t)(t & 1) * HID;
            const _Float16* sx = xv + kbase;
            const _Float16* sh = hv + kbase;

            v8f acc = {};
#pragma unroll 4
            for (int kk = 0; kk < 32; ++kk) {       // K 0..1023 (x part)
                v16h a  = build_a(sx + kk * 32);
                v16h bt = *(const v16h*)(wt + (size_t)kk * 512);   // ds_load
                acc = __builtin_amdgcn_wmma_f32_16x16x32_f16(
                    false, a, false, bt, (short)0, acc, false, false);
            }
#pragma unroll 4
            for (int kk = 0; kk < 32; ++kk) {       // K 1024..2047 (h part)
                v16h a  = build_a(sh + kk * 32);
                v16h bt = *(const v16h*)(wt + (size_t)(32 + kk) * 512);
                acc = __builtin_amdgcn_wmma_f32_16x16x32_f16(
                    false, a, false, bt, (short)0, acc, false, false);
            }

            // D row 0: lanes 0..15 hold the 16 outputs (col n = lane).
            if (lane < 16) gpre[gate][lane] = acc[0];
            __syncthreads();

            if (tid < 16) {
                const int j = j0 + tid;
                float pf = gpre[0][tid] + bfv[j];
                float pi = gpre[1][tid] + biv[j];
                float pg = gpre[2][tid] + bcv[j];
                float po = gpre[3][tid] + bov[j];
                float f = 1.0f / (1.0f + expf(-pf));
                float i = 1.0f / (1.0f + expf(-pi));
                float g = tanhf(pg);
                float o = 1.0f / (1.0f + expf(-po));
                float cn = f * cb[j] + i * g;
                cb[j] = cn;
                float h = o * tanhf(cn);
                if (bIdx == 0) {
                    h0[(size_t)((t + 1) & 1) * HID + j] = (_Float16)h;
                    o0h[(size_t)(t & 1) * HID + j] = (_Float16)o;  // o-gate feeds block1
                } else {
                    h1[(size_t)((t + 1) & 1) * HID + j] = (_Float16)h;
                    if (t == SEQ - 1) o1[j] = o;                   // final o-gate
                }
            }
            if (tid == 0 && bIdx == 0 && t + 1 < SEQ)
                __builtin_prefetch(xh + (size_t)(t + 1) * D_IN, 0, 3);
        }

        // ---- grid-wide sense-reversing barrier (agent scope) ----
        __syncthreads();
        if (tid == 0) {
            unsigned gen = __hip_atomic_load(bar_gen, __ATOMIC_ACQUIRE,
                                             __HIP_MEMORY_SCOPE_AGENT);
            unsigned a = __hip_atomic_fetch_add(bar_cnt, 1u, __ATOMIC_ACQ_REL,
                                                __HIP_MEMORY_SCOPE_AGENT);
            if (a == NWG - 1) {
                __hip_atomic_store(bar_cnt, 0u, __ATOMIC_RELAXED,
                                   __HIP_MEMORY_SCOPE_AGENT);
                __hip_atomic_fetch_add(bar_gen, 1u, __ATOMIC_RELEASE,
                                       __HIP_MEMORY_SCOPE_AGENT);
            } else {
                while (__hip_atomic_load(bar_gen, __ATOMIC_ACQUIRE,
                                         __HIP_MEMORY_SCOPE_AGENT) == gen)
                    __builtin_amdgcn_s_sleep(1);
            }
        }
        __syncthreads();
    }
}

__global__ void classifier(const float* __restrict__ Wcls,
                           const float* __restrict__ bcls,
                           const float* __restrict__ xin,
                           float* __restrict__ logits) {
    int r = blockIdx.x * 128 + threadIdx.x;
    if (r >= NCLS) return;
    const float* wr = Wcls + (size_t)r * HID;
    float s = bcls[r];
#pragma unroll 4
    for (int k = 0; k < HID; ++k) s += wr[k] * xin[k];
    logits[r] = s;
}

__global__ void log_softmax_k(const float* __restrict__ logits,
                              float* __restrict__ out) {
    __shared__ float red[256];
    const int tid = threadIdx.x;
    float m = -INFINITY;
    for (int i = tid; i < NCLS; i += 256) m = fmaxf(m, logits[i]);
    red[tid] = m; __syncthreads();
    for (int s = 128; s > 0; s >>= 1) {
        if (tid < s) red[tid] = fmaxf(red[tid], red[tid + s]);
        __syncthreads();
    }
    const float mx = red[0]; __syncthreads();
    float sum = 0.0f;
    for (int i = tid; i < NCLS; i += 256) sum += expf(logits[i] - mx);
    red[tid] = sum; __syncthreads();
    for (int s = 128; s > 0; s >>= 1) {
        if (tid < s) red[tid] += red[tid + s];
        __syncthreads();
    }
    const float ls = logf(red[0]);
    for (int i = tid; i < NCLS; i += 256) out[i] = logits[i] - mx - ls;
}

extern "C" void kernel_launch(void* const* d_in, const int* in_sizes, int n_in,
                              void* d_out, int out_size, void* d_ws, size_t ws_size,
                              hipStream_t stream) {
    (void)in_sizes; (void)n_in; (void)out_size; (void)ws_size;

    const float* inputs = (const float*)d_in[0];
    const float* Wf  = (const float*)d_in[1];
    const float* bf_ = (const float*)d_in[2];
    const float* Wi  = (const float*)d_in[3];
    const float* bi_ = (const float*)d_in[4];
    const float* Wc  = (const float*)d_in[5];
    const float* bc_ = (const float*)d_in[6];
    const float* Wo  = (const float*)d_in[7];
    const float* bo_ = (const float*)d_in[8];
    const float* Wcls = (const float*)d_in[9];
    const float* bcls = (const float*)d_in[10];

    char* ws = (char*)d_ws;
    size_t off = 0;
    auto carve = [&](size_t bytes) -> char* {
        char* p = ws + off;
        off = (off + bytes + 255) & ~(size_t)255;
        return p;
    };
    _Float16* xh      = (_Float16*)carve((size_t)SEQ * D_IN * 2);          // 16 MB
    _Float16* wpack   = (_Float16*)carve((size_t)2 * 4 * HID * KTOT * 2);  // 32 MB
    _Float16* hstates = (_Float16*)carve((size_t)3 * 2 * HID * 2);         // h0|h1|o0h
    float*    cbuf    = (float*)   carve((size_t)2 * HID * 4);
    float*    o1      = (float*)   carve((size_t)HID * 4);
    float*    lgt     = (float*)   carve((size_t)1024 * 4);
    unsigned* bar     = (unsigned*)carve(256);

    _Float16* h0  = hstates;
    _Float16* h1  = hstates + 2 * HID;
    _Float16* o0h = hstates + 4 * HID;

    pack_weights  <<<65536, 256, 0, stream>>>(Wf, Wi, Wc, Wo, wpack);
    convert_inputs<<<(SEQ * D_IN) / 256, 256, 0, stream>>>(inputs, xh);
    init_state    <<<24, 256, 0, stream>>>(hstates, cbuf, bar);

    const size_t lds_bytes = (size_t)4 * 64 * 512 * sizeof(_Float16);      // 256 KB
    lstm_persistent<<<NWG, 128, lds_bytes, stream>>>(
        xh, wpack, h0, h1, o0h, cbuf, o1,
        bf_, bi_, bc_, bo_, bar, bar + 1);

    classifier   <<<8, 128, 0, stream>>>(Wcls, bcls, o1, lgt);
    log_softmax_k<<<1, 256, 0, stream>>>(lgt, (float*)d_out);
}
